// PiFormerBlock_12206297055406
// MI455X (gfx1250) — matrix operand order repaired and checked by reading, not compiled, not run on hardware
//
#include <hip/hip_runtime.h>
#include <hip/hip_bf16.h>
#include <stdint.h>

// ---------------------------------------------------------------------------
// PiFormer block for MI455X (gfx1250): bf16 WMMA GEMMs + async LDS staging.
// Shapes: x[4,2048,1024], D=1024, H=16, Dh=64, D_FF=4096, tokens M=8192.
// ---------------------------------------------------------------------------

typedef __bf16 bf16;
typedef __attribute__((ext_vector_type(16))) __bf16 v16bf;
typedef __attribute__((ext_vector_type(8)))  __bf16 v8bf;
typedef __attribute__((ext_vector_type(8)))  float  v8f;
typedef __attribute__((ext_vector_type(4)))  int    v4i;

#define DMODEL 1024
#define DFF    4096
#define NHEADS 16
#define DH     64
#define BATCH  4
#define SEQ    2048
#define NTOK   (BATCH*SEQ)

#define BM 128
#define BN 64
#define BK 32
#define PAD 40   // padded LDS row (bf16 units) -> conflict-free ds_load_b128

enum { EPI_F32 = 0, EPI_PHI = 1, EPI_GELU = 2, EPI_ADDRES = 3 };

// ---------------- async global->LDS (CDNA5) with safe fallback -------------
#if defined(__gfx1250__) && __has_builtin(__builtin_amdgcn_global_load_async_to_lds_b128)
#define USE_ASYNC_LDS 1
#else
#define USE_ASYNC_LDS 0
#endif

typedef __attribute__((address_space(1))) v4i* gp128_t;  // global (AS1)
typedef __attribute__((address_space(3))) v4i* lp128_t;  // LDS (AS3)

__device__ __forceinline__ void cp16(const bf16* g, bf16* l) {
#if USE_ASYNC_LDS
  // flat->AS1 is numerically identity; flat->AS3 keeps low 32 bits (LDS offset)
  __builtin_amdgcn_global_load_async_to_lds_b128(
      (gp128_t)(unsigned long long)(uintptr_t)g,
      (lp128_t)(unsigned int)(uintptr_t)l,
      0, 0);
#else
  *(uint4*)l = *(const uint4*)g;
#endif
}

__device__ __forceinline__ void wait_async() {
#if USE_ASYNC_LDS
#if __has_builtin(__builtin_amdgcn_s_wait_asynccnt)
  __builtin_amdgcn_s_wait_asynccnt(0);
#else
  asm volatile("s_wait_asynccnt 0" ::: "memory");
#endif
#endif
}

// ---------------- WMMA helpers ---------------------------------------------
__device__ __forceinline__ v8f wmma_bf16(v16bf a, v16bf b, v8f c) {
  return __builtin_amdgcn_wmma_f32_16x16x32_bf16(
      /*neg_a=*/false, a, /*neg_b=*/false, b,
      /*c_mod=*/(short)0, c, /*reuse_a=*/false, /*reuse_b=*/false);
}

union FragU { v16bf v; v8bf h[2]; };

// A fragment (16x32 bf16): lane&15 = M row; lanes 0-15 take K 0..7,16..23,
// lanes 16-31 take K 8..15,24..31 (per ISA 16-bit A layout).
__device__ __forceinline__ v16bf load_fa(const bf16* sA, int row0, int lane) {
  int m  = row0 + (lane & 15);
  int k0 = (lane >> 4) * 8;
  FragU u;
  u.h[0] = *(const v8bf*)(sA + m * PAD + k0);
  u.h[1] = *(const v8bf*)(sA + m * PAD + k0 + 16);
  return u.v;
}

// B fragment (32x16 bf16) from W^T[N,K]: lane&15 = N col; lanes 0-15 hold
// K 0..7,16..23, lanes 16-31 hold K 8..15,24..31 (matches A layout).
__device__ __forceinline__ v16bf load_fb(const bf16* sB, int col0, int lane) {
  int n  = col0 + (lane & 15);
  int k0 = (lane >> 4) * 8;
  FragU u;
  u.h[0] = *(const v8bf*)(sB + n * PAD + k0);
  u.h[1] = *(const v8bf*)(sB + n * PAD + k0 + 16);
  return u.v;
}

// C/D layout: lanes 0-15: N=lane, M=r; lanes 16-31: N=lane-16, M=8+r
template <int EPI>
__device__ __forceinline__ void store_c(void* Cout, const float* __restrict__ Res,
                                        v8f c, int gm, int gn, int lane, int N) {
  int col   = gn + (lane & 15);
  int rbase = gm + ((lane >> 4) << 3);
#pragma unroll
  for (int r = 0; r < 8; ++r) {
    size_t idx = (size_t)(rbase + r) * N + col;
    float v = c[r];
    if (EPI == EPI_PHI) {
      // elu(x)+1 : x>0 -> x+1 ; x<=0 -> exp(x)
      v = v > 0.f ? v + 1.f : __expf(v);
      ((float*)Cout)[idx] = v;
    } else if (EPI == EPI_GELU) {
      // StructuredLookupActivation: quantize to 256-level grid, tanh-GELU
      float q  = rintf(v * 10.0f);
      q        = fminf(fmaxf(q, -128.0f), 127.0f);
      float xq = q * 0.1f;
      float t  = 0.7978845608028654f * (xq + 0.044715f * xq * xq * xq);
      float g  = 0.5f * xq * (1.0f + tanhf(t));
      ((bf16*)Cout)[idx] = (bf16)g;
    } else if (EPI == EPI_ADDRES) {
      ((float*)Cout)[idx] = Res[idx] + v;
    } else {
      ((float*)Cout)[idx] = v;
    }
  }
}

// ---------------- WMMA GEMM: C[M,N] = A[M,K] (bf16) * Wt[N,K]^T (bf16) -----
template <int EPI>
__global__ __launch_bounds__(256) void gemm_wmma_k(
    const bf16* __restrict__ A, const bf16* __restrict__ Wt,
    void* __restrict__ Cout, const float* __restrict__ Res,
    int M, int N, int K) {
  __shared__ __attribute__((aligned(16))) bf16 As[2][BM * PAD];
  __shared__ __attribute__((aligned(16))) bf16 Bs[2][BN * PAD];

  const int tid  = threadIdx.x;
  const int lane = tid & 31;
  const int wave = tid >> 5;
  const int m0 = blockIdx.x * BM;
  const int n0 = blockIdx.y * BN;
  const int wm = (wave & 3) * 32;   // 4 waves along M (32 rows each)
  const int wn = (wave >> 2) * 32;  // 2 waves along N (32 cols each)

  auto stageA = [&](int buf, int kb) {
    for (int c = tid; c < (BM * BK) / 8; c += 256) {  // 512 16B chunks
      int r = c >> 2, ch = c & 3;
      cp16(A + (size_t)(m0 + r) * K + kb + ch * 8, &As[buf][r * PAD + ch * 8]);
    }
  };
  auto stageB = [&](int buf, int kb) {
    for (int c = tid; c < (BN * BK) / 8; c += 256) {  // 256 16B chunks
      int r = c >> 2, ch = c & 3;
      cp16(Wt + (size_t)(n0 + r) * K + kb + ch * 8, &Bs[buf][r * PAD + ch * 8]);
    }
  };

  v8f c00 = {0.f, 0.f, 0.f, 0.f, 0.f, 0.f, 0.f, 0.f};
  v8f c01 = c00, c10 = c00, c11 = c00;

  stageA(0, 0);
  stageB(0, 0);
  wait_async();
  __syncthreads();

  const int nk = K / BK;
  for (int kb = 0; kb < nk; ++kb) {
    const int cur = kb & 1;
    if (kb + 1 < nk) {  // prefetch next K-slice into the other buffer
      stageA(cur ^ 1, (kb + 1) * BK);
      stageB(cur ^ 1, (kb + 1) * BK);
    }
    const bf16* sA = As[cur];
    const bf16* sB = Bs[cur];
    v16bf a0 = load_fa(sA, wm, lane);
    v16bf a1 = load_fa(sA, wm + 16, lane);
    v16bf b0 = load_fb(sB, wn, lane);
    v16bf b1 = load_fb(sB, wn + 16, lane);
    c00 = wmma_bf16(a0, b0, c00);
    c01 = wmma_bf16(a0, b1, c01);
    c10 = wmma_bf16(a1, b0, c10);
    c11 = wmma_bf16(a1, b1, c11);
    wait_async();
    __syncthreads();
  }

  store_c<EPI>(Cout, Res, c00, m0 + wm,      n0 + wn,      lane, N);
  store_c<EPI>(Cout, Res, c01, m0 + wm,      n0 + wn + 16, lane, N);
  store_c<EPI>(Cout, Res, c10, m0 + wm + 16, n0 + wn,      lane, N);
  store_c<EPI>(Cout, Res, c11, m0 + wm + 16, n0 + wn + 16, lane, N);
}

// ---------------- weight convert + transpose fp32[K,N] -> bf16[N,K] --------
__global__ void cvt_t_k(const float* __restrict__ W, bf16* __restrict__ Wt,
                        int K, int N) {
  size_t i = (size_t)blockIdx.x * blockDim.x + threadIdx.x;
  size_t total = (size_t)K * N;
  if (i >= total) return;
  int k = (int)(i / N), n = (int)(i % N);
  Wt[(size_t)n * K + k] = (bf16)W[i];
}

// ---------------- LayerNorm (D=1024) -> bf16 --------------------------------
__global__ __launch_bounds__(256) void layernorm_k(
    const float* __restrict__ X, const float* __restrict__ G,
    const float* __restrict__ Bb, bf16* __restrict__ O) {
  const int row = blockIdx.x;
  const int tid = threadIdx.x;
  const float* xr = X + (size_t)row * DMODEL;
  float v[4], s = 0.f, s2 = 0.f;
#pragma unroll
  for (int i = 0; i < 4; ++i) {
    v[i] = xr[tid + i * 256];
    s += v[i];
    s2 += v[i] * v[i];
  }
  __shared__ float r1[256], r2[256];
  r1[tid] = s; r2[tid] = s2;
  __syncthreads();
  for (int off = 128; off > 0; off >>= 1) {
    if (tid < off) { r1[tid] += r1[tid + off]; r2[tid] += r2[tid + off]; }
    __syncthreads();
  }
  const float mu  = r1[0] * (1.f / DMODEL);
  const float var = r2[0] * (1.f / DMODEL) - mu * mu;
  const float rs  = rsqrtf(var + 1e-5f);
  bf16* orow = O + (size_t)row * DMODEL;
#pragma unroll
  for (int i = 0; i < 4; ++i) {
    int idx = tid + i * 256;
    orow[idx] = (bf16)((v[i] - mu) * rs * G[idx] + Bb[idx]);
  }
}

// ---------------- linear attention: kv[b,h,d,e], z[b,h,d] -------------------
__global__ __launch_bounds__(256) void kv_z_k(const float* __restrict__ phiK,
                                              const float* __restrict__ V,
                                              float* __restrict__ KV,
                                              float* __restrict__ Z) {
  const int bh = blockIdx.x;          // B*H = 64 blocks
  const int b = bh >> 4, h = bh & 15;
  const int tid = threadIdx.x;
  const int d  = tid >> 2;            // 0..63
  const int e0 = (tid & 3) * 16;      // 16 e's per thread
  const float* pk = phiK + (size_t)b * SEQ * DMODEL + h * DH + d;
  const float* pv = V    + (size_t)b * SEQ * DMODEL + h * DH + e0;
  float acc[16];
#pragma unroll
  for (int i = 0; i < 16; ++i) acc[i] = 0.f;
  float zacc = 0.f;
  for (int s = 0; s < SEQ; ++s) {
    float kk = pk[(size_t)s * DMODEL];
    const float* pvs = pv + (size_t)s * DMODEL;
#pragma unroll
    for (int i = 0; i < 16; ++i) acc[i] += kk * pvs[i];
    zacc += kk;
  }
  float* kvp = KV + (size_t)bh * DH * DH + d * DH + e0;
#pragma unroll
  for (int i = 0; i < 16; ++i) kvp[i] = acc[i];
  if ((tid & 3) == 0) Z[bh * DH + d] = zacc;
}

// out[token, h*64+e] = (phi_q . kv[:,e]) / (phi_q . z + 1e-6) -> bf16
__global__ __launch_bounds__(256) void attn_combine_k(
    const float* __restrict__ phiQ, const float* __restrict__ KV,
    const float* __restrict__ Z, bf16* __restrict__ Out) {
  const int bh = blockIdx.x;          // 64
  const int b = bh >> 4, h = bh & 15;
  const int s0 = blockIdx.y * 64;     // 32 chunks of 64 tokens
  const int tid = threadIdx.x;
  const int e = tid & 63;
  __shared__ float skv[DH * DH];
  __shared__ float sz[DH];
  for (int i = tid; i < DH * DH; i += 256) skv[i] = KV[(size_t)bh * DH * DH + i];
  if (tid < DH) sz[tid] = Z[bh * DH + tid];
  __syncthreads();
  for (int j = 0; j < 16; ++j) {
    int s = s0 + (tid >> 6) + 4 * j;
    const float* pq = phiQ + (size_t)(b * SEQ + s) * DMODEL + h * DH;
    float num = 0.f, den = 0.f;
#pragma unroll 8
    for (int d = 0; d < DH; ++d) {
      float q = pq[d];
      num += q * skv[d * DH + e];
      den += q * sz[d];
    }
    Out[(size_t)(b * SEQ + s) * DMODEL + h * DH + e] = (bf16)(num / (den + 1e-6f));
  }
}

// ---------------------------------------------------------------------------
extern "C" void kernel_launch(void* const* d_in, const int* in_sizes, int n_in,
                              void* d_out, int out_size, void* d_ws, size_t ws_size,
                              hipStream_t stream) {
  const float* x    = (const float*)d_in[0];
  const float* ln1g = (const float*)d_in[1];
  const float* ln1b = (const float*)d_in[2];
  const float* wq   = (const float*)d_in[3];
  const float* wk   = (const float*)d_in[4];
  const float* wv   = (const float*)d_in[5];
  const float* wo   = (const float*)d_in[6];
  const float* ln2g = (const float*)d_in[7];
  const float* ln2b = (const float*)d_in[8];
  const float* w1   = (const float*)d_in[9];
  const float* w2   = (const float*)d_in[10];
  float* out = (float*)d_out;

  // workspace layout (~138 MiB total, fits MI455X 192MB L2 working set)
  const size_t MiB = 1ull << 20;
  char* ws = (char*)d_ws;
  bf16*  wq_t = (bf16*)(ws + 0 * MiB);    // [1024,1024]
  bf16*  wk_t = (bf16*)(ws + 2 * MiB);
  bf16*  wv_t = (bf16*)(ws + 4 * MiB);
  bf16*  wo_t = (bf16*)(ws + 6 * MiB);
  bf16*  w1_t = (bf16*)(ws + 8 * MiB);    // [4096,1024]
  bf16*  w2_t = (bf16*)(ws + 16 * MiB);   // [1024,4096]
  bf16*  xn   = (bf16*)(ws + 24 * MiB);   // [8192,1024] bf16; reused as attn_in
  float* phiq = (float*)(ws + 40 * MiB);  // [8192,1024] f32; region reused as h
  float* phik = (float*)(ws + 72 * MiB);  // [8192,1024] f32; region reused as act
  float* vbuf = (float*)(ws + 104 * MiB); // [8192,1024] f32
  float* kvb  = (float*)(ws + 136 * MiB); // [64,64,64]  f32 (1 MiB)
  float* zb   = (float*)(ws + 137 * MiB); // [64,64]     f32
  bf16*  attn_in = xn;                    // xn dead after v GEMM
  bf16*  h    = (bf16*)(ws + 40 * MiB);   // phiq dead after attn_combine
  bf16*  act  = (bf16*)(ws + 72 * MiB);   // [8192,4096] bf16 over phik+vbuf

  dim3 blk(256);

  // 1) weights -> bf16, transposed to [N,K]
  cvt_t_k<<<(1024 * 1024 + 255) / 256, blk, 0, stream>>>(wq, wq_t, 1024, 1024);
  cvt_t_k<<<(1024 * 1024 + 255) / 256, blk, 0, stream>>>(wk, wk_t, 1024, 1024);
  cvt_t_k<<<(1024 * 1024 + 255) / 256, blk, 0, stream>>>(wv, wv_t, 1024, 1024);
  cvt_t_k<<<(1024 * 1024 + 255) / 256, blk, 0, stream>>>(wo, wo_t, 1024, 1024);
  cvt_t_k<<<(1024 * 4096 + 255) / 256, blk, 0, stream>>>(w1, w1_t, 1024, 4096);
  cvt_t_k<<<(4096 * 1024 + 255) / 256, blk, 0, stream>>>(w2, w2_t, 4096, 1024);

  // 2) LN1 -> bf16
  layernorm_k<<<NTOK, blk, 0, stream>>>(x, ln1g, ln1b, xn);

  // 3) q/k/v projections (phi = elu+1 fused on q,k)
  dim3 g1(NTOK / BM, DMODEL / BN);
  gemm_wmma_k<EPI_PHI><<<g1, blk, 0, stream>>>(xn, wq_t, phiq, nullptr, NTOK, DMODEL, DMODEL);
  gemm_wmma_k<EPI_PHI><<<g1, blk, 0, stream>>>(xn, wk_t, phik, nullptr, NTOK, DMODEL, DMODEL);
  gemm_wmma_k<EPI_F32><<<g1, blk, 0, stream>>>(xn, wv_t, vbuf, nullptr, NTOK, DMODEL, DMODEL);

  // 4) linear attention inner reductions + combine
  kv_z_k<<<BATCH * NHEADS, blk, 0, stream>>>(phik, vbuf, kvb, zb);
  attn_combine_k<<<dim3(BATCH * NHEADS, SEQ / 64), blk, 0, stream>>>(phiq, kvb, zb, attn_in);

  // 5) output projection + residual  (out = x + attn @ wo)
  gemm_wmma_k<EPI_ADDRES><<<g1, blk, 0, stream>>>(attn_in, wo_t, out, x, NTOK, DMODEL, DMODEL);

  // 6) LN2 -> bf16
  layernorm_k<<<NTOK, blk, 0, stream>>>(out, ln2g, ln2b, h);

  // 7) FFN: (lookup-GELU(h @ w1)) @ w2 + residual
  dim3 g2(NTOK / BM, DFF / BN);
  gemm_wmma_k<EPI_GELU><<<g2, blk, 0, stream>>>(h, w1_t, act, nullptr, NTOK, DFF, DMODEL);
  gemm_wmma_k<EPI_ADDRES><<<g1, blk, 0, stream>>>(act, w2_t, out, out, NTOK, DMODEL, DFF);
}
